// GCN_75471165325723
// MI455X (gfx1250) — compile-verified, matrix operand order checked by reading, not run on hardware
//
#include <hip/hip_runtime.h>
#include <hip/hip_bf16.h>

// ---------------------------------------------------------------------------
// CDNA5 (gfx1250) GCN autoencoder.
// GEMMs: v_wmma_f32_16x16x32_bf16, double-buffered async global->LDS A tiles,
// pre-packed bf16 fragment-major weights (one 32B vector load per B frag).
// ---------------------------------------------------------------------------

typedef __attribute__((ext_vector_type(16))) __bf16 v16bf;
typedef __attribute__((ext_vector_type(8)))  float  v8f;
typedef __attribute__((ext_vector_type(4)))  int    v4i;

#define GCN_AS1 __attribute__((address_space(1)))
#define GCN_AS3 __attribute__((address_space(3)))

#if __has_builtin(__builtin_amdgcn_global_load_async_to_lds_b128)
#define GCN_ASYNC_LDS 1
#else
#define GCN_ASYNC_LDS 0
#endif

#if GCN_ASYNC_LDS
#if __has_builtin(__builtin_amdgcn_s_wait_asynccnt)
#define GCN_WAIT_ASYNC(n) __builtin_amdgcn_s_wait_asynccnt(n)
#else
#define GCN_WAIT_ASYNC(n) asm volatile("s_wait_asynccnt %0" ::"i"(n) : "memory")
#endif
#else
#define GCN_WAIT_ASYNC(n) do { } while (0)
#endif

// ---------------------------- small utility kernels ------------------------

__global__ void fill_f32(float* __restrict__ p, float v, long n) {
  long i = (long)blockIdx.x * blockDim.x + threadIdx.x;
  if (i < n) p[i] = v;
}

__global__ void deg_kernel(const long long* __restrict__ dst,
                           float* __restrict__ deg, int E) {
  int e = blockIdx.x * blockDim.x + threadIdx.x;
  if (e < E) atomicAdd(&deg[(int)dst[e]], 1.0f);
}

__global__ void rsqrt_kernel(float* __restrict__ d, int n) {
  int i = blockIdx.x * blockDim.x + threadIdx.x;
  if (i < n) d[i] = rsqrtf(d[i]);   // deg >= 1 always (self-loop)
}

// acc[i] = act(dinv[row]*acc[i] + bias[col]); F = 1<<logF features per row
__global__ void finalize_kernel(float* __restrict__ acc,
                                const float* __restrict__ dinv,
                                const float* __restrict__ bias,
                                long total, int logF, int doRelu) {
  long i = (long)blockIdx.x * blockDim.x + threadIdx.x;
  if (i >= total) return;
  int row = (int)(i >> logF);
  int col = (int)(i & ((1 << logF) - 1));
  float v = dinv[row] * acc[i] + bias[col];
  acc[i] = (doRelu && v < 0.f) ? 0.f : v;
}

// scatter-add: acc[dst[e]][c..c+3] += msg[src[e]][c..c+3]; 1 thread = 1 float4
__global__ void scatter_kernel(const long long* __restrict__ src,
                               const long long* __restrict__ dst,
                               const float* __restrict__ msg,
                               float* __restrict__ acc,
                               long total, int F, int cshift) {
  long gid = (long)blockIdx.x * blockDim.x + threadIdx.x;
  if (gid >= total) return;
  int e = (int)(gid >> cshift);
  int c = (int)(gid & ((1 << cshift) - 1)) * 4;
  int s = (int)src[e];
  int d = (int)dst[e];
  const float4 v = *(const float4*)(msg + (long)s * F + c);
  float* p = acc + (long)d * F + c;
  atomicAdd(p + 0, v.x);
  atomicAdd(p + 1, v.y);
  atomicAdd(p + 2, v.z);
  atomicAdd(p + 3, v.w);
}

// ---------------------- weight packing: fp32 -> bf16 frags ------------------
// Bp layout: [(kt, nt, lane)] -> 16 contiguous bf16 = that lane's WMMA B frag
// (B fragment: N = lane&15, K = kt*32 + (lane>>4)*8 + {0..7, 16..23}).
__global__ void pack_b_kernel(const float* __restrict__ B,
                              __bf16* __restrict__ Bp, int N, int K) {
  int gid = blockIdx.x * blockDim.x + threadIdx.x;
  int nTiles = N >> 4;
  int total = (K >> 5) * nTiles * 32;
  if (gid >= total) return;
  int L  = gid & 31;
  int nt = (gid >> 5) % nTiles;
  int kt = (gid >> 5) / nTiles;
  int n  = nt * 16 + (L & 15);
  int k0 = kt * 32 + (L >> 4) * 8;
  __bf16* out = Bp + (size_t)gid * 16;
#pragma unroll
  for (int e = 0; e < 8; ++e) {
    out[e]     = (__bf16)B[(size_t)(k0 + e) * N + n];
    out[e + 8] = (__bf16)B[(size_t)(k0 + 16 + e) * N + n];
  }
}

// ------------------------------- WMMA GEMM ---------------------------------
// C[M,N] = rowScale[m] * (A[M,K] @ B[K,N]) + bias[n]
// 16 rows per block; each wave owns NT 16x16 col tiles; block covers full N
// (waves*NT*16 == N at the launch site). M%16==0, K%32==0 for all GEMMs here.
template <int NT>
__launch_bounds__(256)
__global__ void wmma_gemm(const float* __restrict__ A,
                          const __bf16* __restrict__ Bp,   // packed bf16 frags
                          float* __restrict__ C,
                          int N, int K,
                          const float* __restrict__ rowScale,
                          const float* __restrict__ bias) {
  __shared__ float As[2][16 * 32];         // double-buffered 16x32 fp32 A tile

  const int tid     = threadIdx.x;
  const int lane    = tid & 31;
  const int wave    = tid >> 5;
  const int rowBase = blockIdx.x * 16;
  const int m       = lane & 15;           // A frag: M = lane&15
  const int kb      = (lane >> 4) * 8;     // K sub-base per half-wave
  const int nLane   = lane & 15;           // C frag: N = lane&15
  const int nTiles  = N >> 4;

  v8f acc[NT];
#pragma unroll
  for (int t = 0; t < NT; ++t)
#pragma unroll
    for (int r = 0; r < 8; ++r) acc[t][r] = 0.f;

  auto stage = [&](int buf, int kk) {
#if GCN_ASYNC_LDS
    if (tid < 128) {                       // 128 lanes x 16B = 16x32 fp32 tile
      const float* g =
          A + (size_t)(rowBase + (tid >> 3)) * K + kk + (tid & 7) * 4;
      __builtin_amdgcn_global_load_async_to_lds_b128(
          (GCN_AS1 v4i*)g, (GCN_AS3 v4i*)&As[buf][tid * 4], 0, 0);
    }
#else
    for (int i = tid; i < 16 * 32; i += blockDim.x)
      As[buf][i] = A[(size_t)(rowBase + (i >> 5)) * K + kk + (i & 31)];
#endif
  };

  const int nIter = K >> 5;
  stage(0, 0);
  for (int it = 0; it < nIter; ++it) {
    const int cur = it & 1;
    if (it + 1 < nIter) {                  // prefetch next tile, keep 1 in flight
      stage(cur ^ 1, (it + 1) << 5);
      GCN_WAIT_ASYNC(1);                   // older (current) tile has landed
    } else {
      GCN_WAIT_ASYNC(0);
    }
    __syncthreads();

    // A fragment: 4x ds_load_b128 + packed f32->bf16 converts
    const float* as = &As[cur][m * 32 + kb];
    const float4 a0 = *(const float4*)(as);
    const float4 a1 = *(const float4*)(as + 4);
    const float4 a2 = *(const float4*)(as + 16);
    const float4 a3 = *(const float4*)(as + 20);
    v16bf a;
    a[0]  = (__bf16)a0.x; a[1]  = (__bf16)a0.y;
    a[2]  = (__bf16)a0.z; a[3]  = (__bf16)a0.w;
    a[4]  = (__bf16)a1.x; a[5]  = (__bf16)a1.y;
    a[6]  = (__bf16)a1.z; a[7]  = (__bf16)a1.w;
    a[8]  = (__bf16)a2.x; a[9]  = (__bf16)a2.y;
    a[10] = (__bf16)a2.z; a[11] = (__bf16)a2.w;
    a[12] = (__bf16)a3.x; a[13] = (__bf16)a3.y;
    a[14] = (__bf16)a3.z; a[15] = (__bf16)a3.w;

    const __bf16* brow = Bp + (size_t)it * nTiles * 32 * 16;
#pragma unroll
    for (int t = 0; t < NT; ++t) {
      const v16bf b =
          *(const v16bf*)(brow + ((size_t)((wave * NT + t) * 32 + lane)) * 16);
      acc[t] = __builtin_amdgcn_wmma_f32_16x16x32_bf16(
          false, a, false, b, (short)0, acc[t], false, false);
    }
    __syncthreads();                       // readers done before buf reuse
  }

  // Epilogue: C/D f32 16x16 layout -> row = rowBase + r + 8*(lane>=16)
  const int roff = (lane >> 4) * 8;
  float scs[8];
#pragma unroll
  for (int r = 0; r < 8; ++r)
    scs[r] = rowScale ? rowScale[rowBase + roff + r] : 1.f;

#pragma unroll
  for (int t = 0; t < NT; ++t) {
    const int col = (wave * NT + t) * 16 + nLane;
    const float bv = bias ? bias[col] : 0.f;
#pragma unroll
    for (int r = 0; r < 8; ++r)
      C[(size_t)(rowBase + roff + r) * N + col] = acc[t][r] * scs[r] + bv;
  }
}

// ------------------------------- launcher ----------------------------------

extern "C" void kernel_launch(void* const* d_in, const int* in_sizes, int n_in,
                              void* d_out, int out_size, void* d_ws,
                              size_t ws_size, hipStream_t stream) {
  (void)n_in; (void)out_size; (void)ws_size;

  const float*     x     = (const float*)d_in[0];
  const long long* edges = (const long long*)d_in[1];  // int64 [2,E] flat
  const float*     W1    = (const float*)d_in[2];
  const float*     b1    = (const float*)d_in[3];
  const float*     W2    = (const float*)d_in[4];
  const float*     b2    = (const float*)d_in[5];
  const float*     Wd    = (const float*)d_in[6];
  const float*     bd    = (const float*)d_in[7];

  const int F = 1024, H = 512, O = 64;
  const int N = in_sizes[0] / F;           // 100000 (divisible by 16)
  const int E = in_sizes[1] / 2;           // 600000
  const long long* src = edges;
  const long long* dst = edges + E;

  // workspace layout (fp32 slots, all carve-outs 256-float aligned)
  float* dinv = (float*)d_ws;                                   // N
  float* hs   = dinv + (((size_t)N + 255) & ~(size_t)255);      // N*H
  float* acc1 = hs + (size_t)N * H;                             // N*H
  float* h2s  = acc1 + (size_t)N * H;                           // N*O
  float* pk   = h2s + (size_t)N * O;                            // packed weights
  __bf16* Bp1 = (__bf16*)pk;                                    // F*H bf16
  __bf16* Bp2 = Bp1 + (size_t)F * H;                            // H*O bf16
  __bf16* Bp3 = Bp2 + (size_t)H * O;                            // O*F bf16

  float* emb   = (float*)d_out;                                 // N*O (out 0)
  float* recon = emb + (size_t)N * O;                           // N*F (out 1)

  // 0) pack weights to bf16 fragment-major
  { int t = (F / 32) * (H / 16) * 32;
    pack_b_kernel<<<(t + 255) / 256, 256, 0, stream>>>(W1, Bp1, H, F); }
  { int t = (H / 32) * (O / 16) * 32;
    pack_b_kernel<<<(t + 255) / 256, 256, 0, stream>>>(W2, Bp2, O, H); }
  { int t = (O / 32) * (F / 16) * 32;
    pack_b_kernel<<<(t + 255) / 256, 256, 0, stream>>>(Wd, Bp3, F, O); }

  // 1) degree (incl. self-loop) -> dinv = rsqrt(deg)
  fill_f32<<<(N + 255) / 256, 256, 0, stream>>>(dinv, 1.0f, N);
  deg_kernel<<<(E + 255) / 256, 256, 0, stream>>>(dst, dinv, E);
  rsqrt_kernel<<<(N + 255) / 256, 256, 0, stream>>>(dinv, N);

  // 2) layer 1: hs = dinv[m] * (x @ W1)   (8 waves * 4 tiles * 16 = 512 = H)
  wmma_gemm<4><<<N / 16, 256, 0, stream>>>(x, Bp1, hs, H, F, dinv, nullptr);

  // 3) aggregate: acc1 = hs (self-loop) + scatter-add over edges
  (void)hipMemcpyAsync(acc1, hs, (size_t)N * H * sizeof(float),
                       hipMemcpyDeviceToDevice, stream);
  {
    long total = (long)E * (H / 4);
    scatter_kernel<<<(unsigned)((total + 255) / 256), 256, 0, stream>>>(
        src, dst, hs, acc1, total, H, 7);
  }
  // 4) h = relu(dinv[d]*acc1 + b1), in place (acc1 becomes h)
  {
    long total = (long)N * H;
    finalize_kernel<<<(unsigned)((total + 255) / 256), 256, 0, stream>>>(
        acc1, dinv, b1, total, 9, 1);
  }

  // 5) layer 2: h2s = dinv[m] * (h @ W2)  (4 waves * 1 tile * 16 = 64 = O)
  wmma_gemm<1><<<N / 16, 128, 0, stream>>>(acc1, Bp2, h2s, O, H, dinv, nullptr);

  // 6) aggregate into d_out embeddings region
  (void)hipMemcpyAsync(emb, h2s, (size_t)N * O * sizeof(float),
                       hipMemcpyDeviceToDevice, stream);
  {
    long total = (long)E * (O / 4);
    scatter_kernel<<<(unsigned)((total + 255) / 256), 256, 0, stream>>>(
        src, dst, h2s, emb, total, O, 4);
  }
  // 7) emb = dinv[d]*acc + b2 (no relu)
  {
    long total = (long)N * O;
    finalize_kernel<<<(unsigned)((total + 255) / 256), 256, 0, stream>>>(
        emb, dinv, b2, total, 6, 0);
  }

  // 8) decoder: recon = emb @ Wd + bd     (8 waves * 8 tiles * 16 = 1024 = F)
  wmma_gemm<8><<<N / 16, 256, 0, stream>>>(emb, Bp3, recon, F, O, nullptr, bd);
}
